// Qwen3MLP_19000935318013
// MI455X (gfx1250) — matrix operand order repaired and checked by reading, not compiled, not run on hardware
//
#include <hip/hip_runtime.h>

typedef __attribute__((ext_vector_type(16))) _Float16 v16h;
typedef __attribute__((ext_vector_type(8)))  float    v8f;
typedef __attribute__((ext_vector_type(4)))  unsigned int v4u;
typedef __attribute__((ext_vector_type(8)))  int      v8i;
typedef __attribute__((ext_vector_type(4)))  int      v4i;

static constexpr int T_ = 4096, H_ = 2048, I_ = 6144;
#define A16_QMAX 32767.0f
#define W4_QMAX  7.0f
#define QEPS     1e-12f

// ---------------- small device helpers ----------------

__device__ __forceinline__ float qdq_val(float v, float scale, float lo, float hi) {
    float q = rintf(v / scale);          // round-half-even, matches jnp.round
    q = fminf(fmaxf(q, lo), hi);
    return q * scale;
}

__device__ __forceinline__ float a16_scale(const float* slot) {
    return fmaxf(*slot / A16_QMAX, QEPS);
}

__device__ __forceinline__ void atomic_absmax(float* slot, float v) {
    // values are >=0 (fabs), so uint bit-pattern ordering == float ordering
    atomicMax((unsigned int*)slot, __float_as_uint(v));
}

__device__ __forceinline__ float wave_max32(float m) {
    #pragma unroll
    for (int off = 16; off > 0; off >>= 1)
        m = fmaxf(m, __shfl_xor(m, off, 32));
    return m;
}

// ---------------- Tensor Data Mover: 2D tile global -> LDS ----------------
// Stages a tile_h x tile_w (elements, 2B each) tile with row stride `stride_elems`
// into LDS at byte offset lds_addr, padding each 64B LDS row (16 DWORDs) with
// +32B (8 DWORDs) to produce a 48-half LDS pitch (bank-friendly, 16B aligned).
__device__ __forceinline__ void tdm_load_tile_f16(unsigned int lds_addr,
                                                  const _Float16* gptr,
                                                  int tile_w, int tile_h,
                                                  int stride_elems) {
    unsigned long long ga = (unsigned long long)(uintptr_t)gptr;
    v4u g0;
    g0.x = 1u;                                             // count=1 (valid), user mode
    g0.y = lds_addr;                                       // lds_addr [63:32]
    g0.z = (unsigned int)(ga & 0xFFFFFFFFu);               // global_addr [95:64]
    g0.w = (unsigned int)((ga >> 32) & 0x01FFFFFFu)        // global_addr [120:96]
         | (2u << 30);                                     // type=2 (image) [127:126]
    v8i g1;
    g1[0] = (1 << 16)        // data_size = 2 bytes
          | (1 << 20)        // pad_enable
          | (3 << 22)        // pad_interval: 16 DWORDs (64B) between pads
          | (7 << 25);       // pad_amount: 8 DWORDs (32B)
    g1[1] = (tile_w & 0xFFFF) << 16;                               // tensor_dim0[15:0]
    g1[2] = ((tile_w >> 16) & 0xFFFF) | ((tile_h & 0xFFFF) << 16); // dim0 hi | tensor_dim1 lo
    g1[3] = ((tile_h >> 16) & 0xFFFF) | ((tile_w & 0xFFFF) << 16); // dim1 hi | tile_dim0
    g1[4] = (tile_h & 0xFFFF);                                     // tile_dim1 (tile_dim2=0)
    g1[5] = stride_elems;                                          // tensor_dim0_stride lo
    g1[6] = 0;                                                     // stride hi | dim1_stride lo
    g1[7] = 0;
    v4i z4 = {0, 0, 0, 0};                                         // groups 2/3 unused (2D)
    v8i z8 = {0, 0, 0, 0, 0, 0, 0, 0};
    __builtin_amdgcn_tensor_load_to_lds(g0, g1, z4, z4, z8, 0);    // 6-arg (clang-23) form
}

// ---------------- scalar slots init ----------------
// slots: [0]=absmax x, [1]=absmax up, [2]=absmax gate, [3]=absmax silu(up), [4]=absmax mid
__global__ void init_scal_kernel(float* scal) {
    if (threadIdx.x < 8) scal[threadIdx.x] = 0.0f;
}

// ---------------- per-tensor absmax of x ----------------
__global__ void absmax_kernel(const float* __restrict__ x, size_t n, float* slot) {
    size_t i = (size_t)blockIdx.x * blockDim.x + threadIdx.x;
    size_t stride = (size_t)gridDim.x * blockDim.x;
    float m = 0.0f;
    for (; i < n; i += stride) m = fmaxf(m, fabsf(x[i]));
    m = wave_max32(m);
    if ((threadIdx.x & 31) == 0) atomic_absmax(slot, m);
}

// ---------------- qdq activations -> f16 ----------------
__global__ void qdq_x_kernel(const float* __restrict__ x, _Float16* __restrict__ out,
                             const float* scal) {
    size_t i = (size_t)blockIdx.x * blockDim.x + threadIdx.x;
    float s = a16_scale(scal);
    out[i] = (_Float16)qdq_val(x[i], s, -A16_QMAX - 1.0f, A16_QMAX);
}

// ---------------- qdq weights (block-32 along input dim) -> f16 ----------------
// one wave32 == one quant block of 32 contiguous elements (rows are multiples of 32)
__global__ void qdq_w_kernel(const float* __restrict__ w, _Float16* __restrict__ out) {
    size_t i = (size_t)blockIdx.x * blockDim.x + threadIdx.x;
    float v = w[i];
    float a = wave_max32(fabsf(v));
    float s = fmaxf(a / W4_QMAX, QEPS);
    out[i] = (_Float16)qdq_val(v, s, -W4_QMAX - 1.0f, W4_QMAX);
}

// ---------------- WMMA GEMM: C[M,N] = A[M,K] * B[N,K]^T ----------------
// 128x128 workgroup tile, 8 waves in 4(M) x 2(N), each wave 32x64 = 2x4 16x16 tiles.
// Staging: each wave TDM-DMAs its 16-row slice of the A and B tiles into LDS,
// double-buffered: tile k+1 DMAs while tile k computes (s_wait_tensorcnt 2 keeps
// exactly the in-flight next tile outstanding; TDM is in-order per wave).
__global__ __launch_bounds__(256)
void gemm_f16_wmma_kernel(const _Float16* __restrict__ A,
                          const _Float16* __restrict__ B,
                          float* __restrict__ C,
                          int M, int N, int K, float* absmax_slot) {
    __shared__ _Float16 sA[2][128][48];   // 32 K-halves + 16-half pad (TDM pad_amount)
    __shared__ _Float16 sB[2][128][48];

    const int tid  = threadIdx.x;
    const int lane = tid & 31;
    const int wid  = tid >> 5;
    const int wm   = wid & 3;          // 0..3 -> 32-row slab
    const int wn   = wid >> 2;         // 0..1 -> 64-col slab
    const int bm   = blockIdx.y;
    const int bn   = blockIdx.x;

    const int r15 = lane & 15;
    const int hi  = lane >> 4;         // 0 or 1

    // wave-uniform staging parameters (SGPR descriptors for TDM)
    const int wid_u = __builtin_amdgcn_readfirstlane(wid);
    const _Float16* gA = A + (size_t)(bm * 128 + wid_u * 16) * K;
    const _Float16* gB = B + (size_t)(bn * 128 + wid_u * 16) * K;
    const unsigned int ldsA[2] = { (unsigned int)(uintptr_t)&sA[0][wid_u * 16][0],
                                   (unsigned int)(uintptr_t)&sA[1][wid_u * 16][0] };
    const unsigned int ldsB[2] = { (unsigned int)(uintptr_t)&sB[0][wid_u * 16][0],
                                   (unsigned int)(uintptr_t)&sB[1][wid_u * 16][0] };

    v8f acc[2][4] = {};

    union Frag { v16h v; int4 i4[2]; };

    // prologue: DMA tile 0 into buffer 0
    tdm_load_tile_f16(ldsA[0], gA, 32, 16, K);
    tdm_load_tile_f16(ldsB[0], gB, 32, 16, K);

    int cur = 0;
    for (int kt = 0; kt < K; kt += 32) {
        const int nxt = cur ^ 1;
        if (kt + 32 < K) {
            // DMA tile k+1 into the other buffer while tile k computes
            tdm_load_tile_f16(ldsA[nxt], gA + kt + 32, 32, 16, K);
            tdm_load_tile_f16(ldsB[nxt], gB + kt + 32, 32, 16, K);
            // warm GL2 two tiles ahead (global_prefetch_b8; speculative, safe past end)
            __builtin_prefetch(gA + kt + 64, 0, 1);
            __builtin_prefetch(gB + kt + 64, 0, 1);
            // wait for tile k's 2 DMAs; leave tile k+1's 2 in flight (in-order per wave)
            __builtin_amdgcn_s_wait_tensorcnt(2);
        } else {
            __builtin_amdgcn_s_wait_tensorcnt(0);
        }
        __syncthreads();

        // A fragments: 16-bit A 16x32 layout (ISA 7.12.2):
        // lanes 0-15: row=lane,  K {0..7, 16..23};  lanes 16-31: row=lane-16, K {8..15, 24..31}
        Frag af[2];
        const int kbA = hi * 8;
        #pragma unroll
        for (int rt = 0; rt < 2; ++rt) {
            int row = wm * 32 + rt * 16 + r15;
            af[rt].i4[0] = *(const int4*)&sA[cur][row][kbA];
            af[rt].i4[1] = *(const int4*)&sA[cur][row][kbA + 16];
        }
        // B fragments: 16-bit B 32x16: lanes 0-15: col=lane, K {0..7, 16..23}; lanes 16-31: col=lane-16, K {8..15, 24..31}
        Frag bf[4];
        const int kbB = hi * 8;
        #pragma unroll
        for (int ct = 0; ct < 4; ++ct) {
            int n = wn * 64 + ct * 16 + r15;
            bf[ct].i4[0] = *(const int4*)&sB[cur][n][kbB];
            bf[ct].i4[1] = *(const int4*)&sB[cur][n][kbB + 16];
        }

        #pragma unroll
        for (int rt = 0; rt < 2; ++rt)
            #pragma unroll
            for (int ct = 0; ct < 4; ++ct)
                acc[rt][ct] = __builtin_amdgcn_wmma_f32_16x16x32_f16(
                    false, af[rt].v, false, bf[ct].v,
                    (short)0, acc[rt][ct], false, false);

        __syncthreads();   // all waves done reading buf `cur` before it is re-DMAed
        cur = nxt;
    }

    // epilogue: C/D layout — VGPR i: lanes 0-15 -> (M=i, N=lane); lanes 16-31 -> (M=i+8, N=lane-16)
    float m = 0.0f;
    #pragma unroll
    for (int rt = 0; rt < 2; ++rt) {
        #pragma unroll
        for (int ct = 0; ct < 4; ++ct) {
            int row0 = bm * 128 + wm * 32 + rt * 16 + hi * 8;
            int col  = bn * 128 + wn * 64 + ct * 16 + r15;
            #pragma unroll
            for (int i = 0; i < 8; ++i) {
                float v = acc[rt][ct][i];
                C[(size_t)(row0 + i) * N + col] = v;
                m = fmaxf(m, fabsf(v));
            }
        }
    }
    if (absmax_slot) {
        m = wave_max32(m);
        if (lane == 0) atomic_absmax(absmax_slot, m);
    }
}

// ---------------- elementwise passes over [T, I] ----------------

// pass 1: absmax of silu(qdq(up))
__global__ void silu_absmax_kernel(const float* __restrict__ up_raw, const float* scal,
                                   float* slot) {
    size_t i = (size_t)blockIdx.x * blockDim.x + threadIdx.x;
    float s_up = a16_scale(scal + 1);
    float u = qdq_val(up_raw[i], s_up, -A16_QMAX - 1.0f, A16_QMAX);
    float act = u / (1.0f + expf(-u));          // silu
    float m = wave_max32(fabsf(act));
    if ((threadIdx.x & 31) == 0) atomic_absmax(slot, m);
}

// pass 2: mid = qdq(gate) * qdq(silu(qdq(up))); overwrite up_raw with mid; absmax(mid)
__global__ void mid_kernel(float* __restrict__ up_raw, const float* __restrict__ gate_raw,
                           float* scal) {
    size_t i = (size_t)blockIdx.x * blockDim.x + threadIdx.x;
    float s_up   = a16_scale(scal + 1);
    float s_gate = a16_scale(scal + 2);
    float s_act  = a16_scale(scal + 3);
    float u = qdq_val(up_raw[i],   s_up,   -A16_QMAX - 1.0f, A16_QMAX);
    float g = qdq_val(gate_raw[i], s_gate, -A16_QMAX - 1.0f, A16_QMAX);
    float a = qdq_val(u / (1.0f + expf(-u)), s_act, -A16_QMAX - 1.0f, A16_QMAX);
    float mid = g * a;
    up_raw[i] = mid;
    float m = wave_max32(fabsf(mid));
    if ((threadIdx.x & 31) == 0) atomic_absmax(scal + 4, m);
}

// pass 3: mid f32 -> qdq -> f16
__global__ void qdq_mid_kernel(const float* __restrict__ mid_raw, _Float16* __restrict__ out,
                               const float* scal) {
    size_t i = (size_t)blockIdx.x * blockDim.x + threadIdx.x;
    float s = a16_scale(scal + 4);
    out[i] = (_Float16)qdq_val(mid_raw[i], s, -A16_QMAX - 1.0f, A16_QMAX);
}

// ---------------- host launcher ----------------

extern "C" void kernel_launch(void* const* d_in, const int* in_sizes, int n_in,
                              void* d_out, int out_size, void* d_ws, size_t ws_size,
                              hipStream_t stream) {
    (void)in_sizes; (void)n_in; (void)out_size; (void)ws_size;

    const float* x      = (const float*)d_in[0];
    const float* w_gate = (const float*)d_in[1];
    const float* w_up   = (const float*)d_in[2];
    const float* w_down = (const float*)d_in[3];
    float* out = (float*)d_out;

    // workspace carve-up (256B aligned)
    char* p = (char*)d_ws;
    auto take = [&](size_t bytes) -> void* {
        void* r = (void*)p;
        p += (bytes + 255) & ~(size_t)255;
        return r;
    };
    float*    scal  = (float*)   take(64);
    _Float16* xh    = (_Float16*)take((size_t)T_ * H_ * 2);
    _Float16* wupq  = (_Float16*)take((size_t)I_ * H_ * 2);
    _Float16* wgq   = (_Float16*)take((size_t)I_ * H_ * 2);
    _Float16* wdq   = (_Float16*)take((size_t)H_ * I_ * 2);
    float*    upr   = (float*)   take((size_t)T_ * I_ * 4);
    float*    gater = (float*)   take((size_t)T_ * I_ * 4);
    _Float16* midh  = (_Float16*)take((size_t)T_ * I_ * 2);

    const size_t nTH = (size_t)T_ * H_;      // 8.39M
    const size_t nTI = (size_t)T_ * I_;      // 25.2M
    const size_t nIH = (size_t)I_ * H_;      // 12.6M

    // 0. reset quantizer state (deterministic per call)
    init_scal_kernel<<<1, 64, 0, stream>>>(scal);

    // 1. per-tensor absmax of x, then qdq(x) -> f16
    absmax_kernel<<<2048, 256, 0, stream>>>(x, nTH, scal + 0);
    qdq_x_kernel<<<(int)(nTH / 256), 256, 0, stream>>>(x, xh, scal);

    // 2. LPBQ weight qdq -> f16 (wave32 == one 32-element block)
    qdq_w_kernel<<<(int)(nIH / 256), 256, 0, stream>>>(w_up,   wupq);
    qdq_w_kernel<<<(int)(nIH / 256), 256, 0, stream>>>(w_gate, wgq);
    qdq_w_kernel<<<(int)(nIH / 256), 256, 0, stream>>>(w_down, wdq);

    // 3. up/gate GEMMs with fused absmax
    dim3 gUp(I_ / 128, T_ / 128);
    gemm_f16_wmma_kernel<<<gUp, 256, 0, stream>>>(xh, wupq, upr,   T_, I_, H_, scal + 1);
    gemm_f16_wmma_kernel<<<gUp, 256, 0, stream>>>(xh, wgq,  gater, T_, I_, H_, scal + 2);

    // 4. silu absmax, mid = qdq(gate)*qdq(silu(qdq(up))), qdq(mid) -> f16
    silu_absmax_kernel<<<(int)(nTI / 256), 256, 0, stream>>>(upr, scal, scal + 3);
    mid_kernel<<<(int)(nTI / 256), 256, 0, stream>>>(upr, gater, scal);
    qdq_mid_kernel<<<(int)(nTI / 256), 256, 0, stream>>>(upr, midh, scal);

    // 5. down GEMM -> out (f32), no absmax needed
    dim3 gDn(H_ / 128, T_ / 128);
    gemm_f16_wmma_kernel<<<gDn, 256, 0, stream>>>(midh, wdq, out, T_, H_, I_, nullptr);
}
